// MulticlassDecorrelatedFocalLoss_70824010711274
// MI455X (gfx1250) — compile-verified, matrix operand order checked by reading, not run on hardware
//
#include <hip/hip_runtime.h>
#include <hip/hip_bf16.h>

// MulticlassDecorrelatedFocalLoss for MI455X (gfx1250, wave32).
//
// Strategy (see analysis): disco collapses to O(N) moments plus two O(N^2)
// passes (davg row-sums, and the cross term T1 = sum nw_i nw_j |dx||de|).
// All operand arrays are staged once into LDS (96KB << 320KB/WGP); packed
// f16 VALU computes the pairwise |diff|*nw products (2 pairs/op) and
// V_WMMA_F32_16X16X32_F16 with B=ones performs the row-sum accumulation in
// f32 on the matrix pipe, co-executing with VALU (WMMA is TRANS-class on
// CDNA5). No N^2 matrices ever touch memory.

typedef _Float16 h16;
typedef _Float16 h2   __attribute__((ext_vector_type(2)));
typedef _Float16 v16h __attribute__((ext_vector_type(16)));
typedef float    v8f  __attribute__((ext_vector_type(8)));

#define NCLS 4          // fixed by reference
#define FOCAL_LAM 0.1f

// ws float "misc" slots:
// 0:sumw 1:focalSum 2..5:S1_c 6:S1_e 7..10:S2_c 11:S2_e 12..15:T1_c(atomic)

// ---------------------------------------------------------------- O(N) prep
__global__ __launch_bounds__(256)
void disco_prep(const float* __restrict__ outs, const int* __restrict__ labels,
                const float* __restrict__ event, const float* __restrict__ w,
                float* __restrict__ misc, float* __restrict__ nwf,
                h16* __restrict__ h, int N) {
  const int t = threadIdx.x;
  __shared__ float red[256];
  __shared__ float s_sumw;

  // phase 1: sum of weights
  float lw = 0.f;
  for (int i = t; i < N; i += 256) lw += w[i];
  red[t] = lw; __syncthreads();
  for (int s = 128; s > 0; s >>= 1) { if (t < s) red[t] += red[t + s]; __syncthreads(); }
  if (t == 0) {
    s_sumw = red[0]; misc[0] = red[0];
    misc[12] = misc[13] = misc[14] = misc[15] = 0.f;   // zero T1 atomics every call
  }
  __syncthreads();
  const float scale = (float)N / s_sumw;

  // phase 2: softmax/probs/focal + weighted moments, f16 operand arrays
  float acc[11];
  #pragma unroll
  for (int q = 0; q < 11; ++q) acc[q] = 0.f;
  for (int i = t; i < N; i += 256) {
    float x[NCLS];
    #pragma unroll
    for (int c = 0; c < NCLS; ++c) x[c] = outs[i * NCLS + c];
    float mx = x[0];
    #pragma unroll
    for (int c = 1; c < NCLS; ++c) mx = fmaxf(mx, x[c]);
    float ex[NCLS], se = 0.f;
    #pragma unroll
    for (int c = 0; c < NCLS; ++c) { ex[c] = __expf(x[c] - mx); se += ex[c]; }
    const float logZ = mx + __logf(se);
    const float inv_se = 1.f / se;

    const float wi = w[i];
    const float nw = wi * scale;
    nwf[i] = nw;
    h[5 * N + i] = (h16)nw;
    const float ev = event[i];
    h[4 * N + i] = (h16)ev;
    acc[5]  += nw * ev;
    acc[10] += nw * ev * ev;
    #pragma unroll
    for (int c = 0; c < NCLS; ++c) {
      const float p = ex[c] * inv_se;
      h[c * N + i] = (h16)p;
      acc[1 + c] += nw * p;        // S1_c
      acc[6 + c] += nw * p * p;    // S2_c
    }
    const int lab = labels[i];
    const float logpt = x[lab] - logZ;
    const float pt = __expf(logpt);
    acc[0] += -(1.f - pt) * (1.f - pt) * logpt * wi;   // GAMMA = 2
  }
  for (int q = 0; q < 11; ++q) {
    red[t] = acc[q]; __syncthreads();
    for (int s = 128; s > 0; s >>= 1) { if (t < s) red[t] += red[t + s]; __syncthreads(); }
    if (t == 0) misc[1 + q] = red[0];
    __syncthreads();
  }
}

// ---------------------------------------------- O(N^2): davg via WMMA rowsum
__global__ __launch_bounds__(256)
void disco_davg(const h16* __restrict__ hg, float* __restrict__ alpha, int N) {
  extern __shared__ char smem[];
  h16* sh = (h16*)smem;                         // 6*N halves: p0..p3, e, nw
  const int t = threadIdx.x;
  {
    const unsigned* src = (const unsigned*)hg;
    unsigned* dst = (unsigned*)sh;
    for (int idx = t; idx < 3 * N; idx += 256) dst[idx] = src[idx];
  }
  __syncthreads();

  const int lane = t & 31;
  const int tile = blockIdx.x * 8 + (t >> 5);   // one 16-row tile per wave
  const int i0 = tile * 16;
  const int i  = i0 + (lane & 15);
  const int kb = (lane >> 4) ? 4 : 0;           // K-half offset, in half2 units
  const h2* nw2 = (const h2*)(sh + 5 * N);
  const float invN = 1.0f / (float)N;

  union { h16 s[16]; v16h v; } Bones;
  #pragma unroll
  for (int q = 0; q < 16; ++q) Bones.s[q] = (h16)1.0f;

  for (int a = 0; a < 5; ++a) {                 // 4 class prob columns + event
    const h16* y = sh + a * N;
    const h2* y2 = (const h2*)y;
    const h16 yi = y[i];
    const h2 yi2 = { yi, yi };
    v8f acc = {};
    for (int jb = 0; jb < N; jb += 32) {
      const int b0 = (jb >> 1) + kb;            // K = kbase .. kbase+7
      const int b1 = b0 + 8;                    // K = 16+kbase .. 16+kbase+7
      union { h2 h[8]; v16h v; } A;
      #pragma unroll
      for (int q = 0; q < 4; ++q) {
        A.h[q]     = __builtin_elementwise_abs(yi2 - y2[b0 + q]) * nw2[b0 + q];
        A.h[4 + q] = __builtin_elementwise_abs(yi2 - y2[b1 + q]) * nw2[b1 + q];
      }
      acc = __builtin_amdgcn_wmma_f32_16x16x32_f16(
          false, A.v, false, Bones.v, (short)0, acc, false, false);
    }
    // D[m,n] is n-independent: lanes 0..15 hold rows r..r+0, lanes 16..31 rows 8+r
    if (lane == 0) {
      #pragma unroll
      for (int r = 0; r < 8; ++r) alpha[a * N + i0 + r] = acc[r] * invN;
    }
    if (lane == 16) {
      #pragma unroll
      for (int r = 0; r < 8; ++r) alpha[a * N + i0 + 8 + r] = acc[r] * invN;
    }
  }
}

// ------------------------------- O(N^2): T1_c = sum nw_i nw_j |dp_c||de|
__global__ __launch_bounds__(256)
void disco_t1(const h16* __restrict__ hg, const float* __restrict__ nwf,
              float* __restrict__ misc, int N) {
  extern __shared__ char smem[];
  h16* sh = (h16*)smem;
  const int t = threadIdx.x;
  {
    const unsigned* src = (const unsigned*)hg;
    unsigned* dst = (unsigned*)sh;
    for (int idx = t; idx < 3 * N; idx += 256) dst[idx] = src[idx];
  }
  __syncthreads();

  const int lane = t & 31;
  const int tile = blockIdx.x * 8 + (t >> 5);
  const int i0 = tile * 16;
  const int i  = i0 + (lane & 15);
  const int kb = (lane >> 4) ? 4 : 0;
  const h2* e2  = (const h2*)(sh + 4 * N);
  const h2* nw2 = (const h2*)(sh + 5 * N);

  const h16 ei = (sh + 4 * N)[i];
  const h2 ei2 = { ei, ei };
  h2 pi2[NCLS];
  #pragma unroll
  for (int c = 0; c < NCLS; ++c) { h16 p = sh[c * N + i]; pi2[c] = (h2){ p, p }; }

  union { h16 s[16]; v16h v; } Bones;
  #pragma unroll
  for (int q = 0; q < 16; ++q) Bones.s[q] = (h16)1.0f;

  v8f accs[NCLS] = {};
  for (int jb = 0; jb < N; jb += 32) {
    const int b0 = (jb >> 1) + kb;
    const int b1 = b0 + 8;
    h2 u[8];                                     // |de| * nw_j
    #pragma unroll
    for (int q = 0; q < 4; ++q) {
      u[q]     = __builtin_elementwise_abs(ei2 - e2[b0 + q]) * nw2[b0 + q];
      u[4 + q] = __builtin_elementwise_abs(ei2 - e2[b1 + q]) * nw2[b1 + q];
    }
    #pragma unroll
    for (int c = 0; c < NCLS; ++c) {
      const h2* p2 = (const h2*)(sh + c * N);
      union { h2 h[8]; v16h v; } A;
      #pragma unroll
      for (int q = 0; q < 4; ++q) {
        A.h[q]     = __builtin_elementwise_abs(pi2[c] - p2[b0 + q]) * u[q];
        A.h[4 + q] = __builtin_elementwise_abs(pi2[c] - p2[b1 + q]) * u[4 + q];
      }
      accs[c] = __builtin_amdgcn_wmma_f32_16x16x32_f16(
          false, A.v, false, Bones.v, (short)0, accs[c], false, false);
    }
  }
  // apply row weight nw_i in f32 and reduce the tile
  const int rbase = i0 + ((lane >> 4) ? 8 : 0);
  #pragma unroll
  for (int c = 0; c < NCLS; ++c) {
    float p = 0.f;
    #pragma unroll
    for (int r = 0; r < 8; ++r) p += accs[c][r] * nwf[rbase + r];
    p += __shfl_xor(p, 16, 32);                  // combine the two lane halves
    if (lane == 0) atomicAdd(&misc[12 + c], p);
  }
}

// ---------------------------------------------------------------- O(N) finish
__global__ __launch_bounds__(256)
void disco_finish(const float* __restrict__ alpha, const float* __restrict__ nwf,
                  const float* __restrict__ misc, float* __restrict__ out, int N) {
  const int t = threadIdx.x;
  __shared__ float red[256];
  __shared__ float res[14];
  float acc[14];
  #pragma unroll
  for (int q = 0; q < 14; ++q) acc[q] = 0.f;
  for (int i = t; i < N; i += 256) {
    const float nw = nwf[i];
    const float ae = alpha[4 * N + i];
    #pragma unroll
    for (int a = 0; a < 5; ++a) {
      const float al = alpha[a * N + i];
      acc[a]     += nw * al;          // N*m_a
      acc[5 + a] += nw * al * al;     // Q_a
    }
    #pragma unroll
    for (int c = 0; c < NCLS; ++c) acc[10 + c] += nw * alpha[c * N + i] * ae;  // R_c
  }
  for (int q = 0; q < 14; ++q) {
    red[t] = acc[q]; __syncthreads();
    for (int s = 128; s > 0; s >>= 1) { if (t < s) red[t] += red[t + s]; __syncthreads(); }
    if (t == 0) res[q] = red[0];
    __syncthreads();
  }
  if (t == 0) {
    const float fN = (float)N, invN = 1.f / fN;
    float m[5], Q[5];
    #pragma unroll
    for (int a = 0; a < 5; ++a) { m[a] = res[a] * invN; Q[a] = res[5 + a]; }
    const float S1e = misc[6], S2e = misc[11];
    const float Vee = (2.f * fN * S2e - 2.f * S1e * S1e) * invN * invN;
    const float BB  = Vee - 2.f * Q[4] * invN + m[4] * m[4];
    float disco = 0.f;
    #pragma unroll
    for (int c = 0; c < NCLS; ++c) {
      const float S1 = misc[2 + c], S2 = misc[7 + c];
      const float V  = (2.f * fN * S2 - 2.f * S1 * S1) * invN * invN;
      const float AA = V - 2.f * Q[c] * invN + m[c] * m[c];
      const float num = misc[12 + c] * invN * invN - 2.f * invN * res[10 + c] + m[c] * m[4];
      disco += num * rsqrtf(AA * BB);
    }
    disco *= (1.f / (float)NCLS);
    const float f = misc[1] * invN;
    out[0] = f; out[1] = disco; out[2] = f + FOCAL_LAM * disco;
  }
}

extern "C" void kernel_launch(void* const* d_in, const int* in_sizes, int n_in,
                              void* d_out, int out_size, void* d_ws, size_t ws_size,
                              hipStream_t stream) {
  const float* outs  = (const float*)d_in[0];
  const int*   labs  = (const int*)d_in[1];
  const float* event = (const float*)d_in[2];
  const float* w     = (const float*)d_in[3];
  float* out = (float*)d_out;
  const int N = in_sizes[2];                    // 8192 (multiple of 128)

  char* ws = (char*)d_ws;
  float* misc  = (float*)ws;                             // 32 f32
  float* nwf   = (float*)(ws + 128);                     // N f32
  float* alpha = (float*)(ws + 128 + (size_t)N * 4);     // 5*N f32
  h16*   h     = (h16*)  (ws + 128 + (size_t)N * 24);    // 6*N f16

  const size_t smem = (size_t)6 * N * sizeof(h16);       // 96 KB @ N=8192
  const int rowBlocks = N / 128;                         // 8 waves * 16 rows

  disco_prep  <<<1,         256, 0,    stream>>>(outs, labs, event, w, misc, nwf, h, N);
  disco_davg  <<<rowBlocks, 256, smem, stream>>>(h, alpha, N);
  disco_t1    <<<rowBlocks, 256, smem, stream>>>(h, nwf, misc, N);
  disco_finish<<<1,         256, 0,    stream>>>(alpha, nwf, misc, out, N);
}
